// Model_51084341018990
// MI455X (gfx1250) — matrix operand order repaired and checked by reading, not compiled, not run on hardware
//
#include <hip/hip_runtime.h>

typedef __bf16 v16bf __attribute__((ext_vector_type(16)));
typedef __bf16 v8bf  __attribute__((ext_vector_type(8)));
typedef float  v8f   __attribute__((ext_vector_type(8)));

#define NN 2000   // nodes
#define DD 32     // D_IN == D_OUT
#define HH 64     // hidden
#define EE 16     // env features
#define PP 24     // periods
#define BB 32     // batch

#define TPAD 72   // T-tile row pitch in bf16 (144 B, 16B-aligned, bank-spread)
#define APAD 40   // A-tile / support-tile row pitch in bf16 (80 B)

__device__ __forceinline__ v16bf cat8(v8bf lo, v8bf hi) {
  return __builtin_shufflevector(lo, hi, 0, 1, 2, 3, 4, 5, 6, 7,
                                         8, 9, 10, 11, 12, 13, 14, 15);
}

// Direct global -> LDS async copy (16 B per active lane), tracked by ASYNCcnt.
// LDS operand is the wave-relative LDS byte address = low 32 bits of the
// generic pointer (flat-aperture truncation rule, ISA 10.2).
__device__ __forceinline__ void async_copy16(void* lds, const void* gaddr) {
  const unsigned ldsOff = (unsigned)(size_t)lds;
  asm volatile("global_load_async_to_lds_b128 %0, %1, off"
               :: "v"(ldsOff), "v"(gaddr) : "memory");
}
__device__ __forceinline__ void wait_async0() {
  asm volatile("s_wait_asynccnt 0x0" ::: "memory");
}

// ---------------------------------------------------------------------------
// Kernel 1: env = relu(env_features @ W_env + b_env); S/T = bf16(src/tgt + env)
// ---------------------------------------------------------------------------
__global__ void k_prep(const float* __restrict__ envf, const float* __restrict__ W_env,
                       const float* __restrict__ b_env, const float* __restrict__ src,
                       const float* __restrict__ tgt,
                       unsigned short* __restrict__ Sst_u, unsigned short* __restrict__ Tst_u) {
  __bf16* Sst = (__bf16*)Sst_u;
  __bf16* Tst = (__bf16*)Tst_u;
  const int n = blockIdx.x;   // 0..NN-1
  const int h = threadIdx.x;  // 0..HH-1
  float acc = b_env[h];
  #pragma unroll
  for (int e = 0; e < EE; ++e) acc += envf[n * EE + e] * W_env[e * HH + h];
  const float env = fmaxf(acc, 0.f);
  for (int p = 0; p < PP; ++p) {
    const int idx = (p * NN + n) * HH + h;
    Sst[idx] = (__bf16)(src[idx] + env);
    Tst[idx] = (__bf16)(tgt[idx] + env);
  }
}

// ---------------------------------------------------------------------------
// Kernel 2: support^T[b][m][k] = bf16(input @ weight); residual = relu(in@W_res+b)
// ---------------------------------------------------------------------------
__global__ void k_support(const float* __restrict__ x, const float* __restrict__ W,
                          const float* __restrict__ Wr, const float* __restrict__ br,
                          unsigned short* __restrict__ supT_u, float* __restrict__ resid) {
  __bf16* supT = (__bf16*)supT_u;
  const int idx = blockIdx.x * blockDim.x + threadIdx.x;  // over BB*NN*DD
  if (idx >= BB * NN * DD) return;
  const int m   = idx & (DD - 1);
  const int row = idx >> 5;        // b*NN + n
  const int b   = row / NN;
  const int n   = row - b * NN;
  const float* xr = x + (size_t)row * DD;
  float s = 0.f, r = br[m];
  #pragma unroll
  for (int d = 0; d < DD; ++d) {
    const float xv = xr[d];
    s += xv * W[d * DD + m];
    r += xv * Wr[d * DD + m];
  }
  supT[((size_t)(b * DD + m)) * NN + n] = (__bf16)s;
  resid[idx] = fmaxf(r, 0.f);
}

// ---------------------------------------------------------------------------
// Issue one k-tile's async copies (T: 32x64 bf16 = 256 x 16B chunks,
// support: 32x32 bf16 = 128 x 16B chunks). Lanes whose chunk starts at
// k >= NN simply skip: stale-but-finite LDS data there is multiplied by
// A-tile entries that the adj-clamp forces to 0, so it never contributes.
// ---------------------------------------------------------------------------
__device__ __forceinline__ void stage_async(__bf16 (*Tl)[TPAD], __bf16 (*Sl)[APAD],
                                            const __bf16* __restrict__ Tbase,
                                            const __bf16* __restrict__ Supb,
                                            int kb, int tid) {
  #pragma unroll
  for (int c = tid; c < 256; c += 128) {
    const int kk = c >> 3, q = c & 7, kg = kb + kk;
    if (kg < NN) async_copy16(&Tl[kk][q * 8], Tbase + (size_t)kg * HH + q * 8);
  }
  const int m = tid >> 2, q = tid & 3;
  if (kb + q * 8 < NN)
    async_copy16(&Sl[m][q * 8], Supb + (size_t)m * NN + kb + q * 8);
}

// ---------------------------------------------------------------------------
// Kernel 3: fused  conv[b] = (relu(S_p T_p^T) * adj) @ support[b] + bias,
//           out = relu(conv + residual).   Never materializes A (384 MB).
// Block = 128 threads (4 wave32), 64 n-rows of one batch. LDS tiles are
// double-buffered and filled by async global->LDS copies issued one k-step
// ahead; 6x v_wmma_f32_16x16x32_bf16 per wave per 32-wide k-step.
// ---------------------------------------------------------------------------
__global__ void __launch_bounds__(128)
k_fused(const int* __restrict__ cyc, const float* __restrict__ adj,
        const unsigned short* __restrict__ Sst_u, const unsigned short* __restrict__ Tst_u,
        const unsigned short* __restrict__ supT_u, const float* __restrict__ resid,
        const float* __restrict__ bias, float* __restrict__ out) {
  const __bf16* Sst  = (const __bf16*)Sst_u;
  const __bf16* Tst  = (const __bf16*)Tst_u;
  const __bf16* supT = (const __bf16*)supT_u;

  __shared__ alignas(16) __bf16 TldsD[2][32][TPAD];   // 9.2 KB  (T_p k-tiles)
  __shared__ alignas(16) __bf16 SupD[2][32][APAD];    // 5.1 KB  (support [m][k])
  __shared__ alignas(16) __bf16 Alds[4][16][APAD];    // 5.1 KB  (per-wave A tile)

  const int b  = blockIdx.y;
  const int nb = blockIdx.x * 64;
  const int p  = cyc[b] % PP;

  const int tid   = threadIdx.x;
  const int wv    = tid >> 5;
  const int lane  = tid & 31;
  const int lcol  = lane & 15;    // N index (B/C/D) == M row (A operand)
  const int lhalf = lane >> 4;
  const int baseK = lhalf * 16;   // A-operand K base (contiguous k = 16h + s, matches B)
  const int kbh   = lhalf * 16;   // B-operand K base (contiguous k = 16h + s, matches A)

  const __bf16* Tbase = Tst + (size_t)p * NN * HH;
  const __bf16* Supb  = supT + (size_t)b * DD * NN;

  // --- Preload this wave's S rows as two A-operand fragments (h 0..31, 32..63)
  const int  myrow = nb + wv * 16 + lcol;
  const int  rowc  = myrow < NN ? myrow : (NN - 1);
  const __bf16* Srow = Sst + ((size_t)p * NN + rowc) * HH;
  const v16bf aS0 = cat8(*(const v8bf*)(Srow + baseK),
                         *(const v8bf*)(Srow + 8 + baseK));
  const v16bf aS1 = cat8(*(const v8bf*)(Srow + 32 + baseK),
                         *(const v8bf*)(Srow + 40 + baseK));

  v8f convAcc0 = {}, convAcc1 = {};

  // --- prologue: kick off tile 0 into buffer 0
  stage_async(TldsD[0], SupD[0], Tbase, Supb, 0, tid);

  const int KITERS = (NN + 31) / 32;  // 63
  for (int kit = 0; kit < KITERS; ++kit) {
    const int kb  = kit * 32;
    const int cur = kit & 1;
    wait_async0();      // this wave's copies into buf[cur] are done
    __syncthreads();    // all waves done; prior reads of buf[!cur] retired
    if (kit + 1 < KITERS)
      stage_async(TldsD[cur ^ 1], SupD[cur ^ 1], Tbase, Supb, kb + 32, tid);

    __bf16 (*Tlds)[TPAD] = TldsD[cur];
    __bf16 (*SupT_l)[APAD] = SupD[cur];

    // ---- A tile: 16(n) x 32(k) = relu(S @ T^T) * adj, two 16-col halves
    #pragma unroll
    for (int koff = 0; koff < 2; ++koff) {
      const int kloc = koff * 16 + lcol;  // local k row = B-operand column
      const v16bf bT0 = cat8(*(const v8bf*)&Tlds[kloc][kbh],
                             *(const v8bf*)&Tlds[kloc][kbh + 8]);
      const v16bf bT1 = cat8(*(const v8bf*)&Tlds[kloc][32 + kbh],
                             *(const v8bf*)&Tlds[kloc][40 + kbh]);
      v8f acc = {};
      acc = __builtin_amdgcn_wmma_f32_16x16x32_bf16(false, aS0, false, bT0,
                                                    (short)0, acc, false, false);
      acc = __builtin_amdgcn_wmma_f32_16x16x32_bf16(false, aS1, false, bT1,
                                                    (short)0, acc, false, false);
      const int kg = kb + koff * 16 + lcol;
      #pragma unroll
      for (int r = 0; r < 8; ++r) {
        const int M = r + 8 * lhalf;
        const int n = nb + wv * 16 + M;
        const float av = fmaxf(acc[r], 0.f);
        const float w  = (n < NN && kg < NN) ? adj[(size_t)n * NN + kg] : 0.f;
        Alds[wv][M][koff * 16 + lcol] = (__bf16)(av * w);
      }
    }
    __syncthreads();  // (same-wave DS is in-order; barrier kept for safety)

    // ---- re-gather A tile as a WMMA A-operand (16 x 32)
    const v16bf a2 = cat8(*(const v8bf*)&Alds[wv][lcol][baseK],
                          *(const v8bf*)&Alds[wv][lcol][8 + baseK]);
    // ---- conv += A_tile @ support_tile  (two 16-wide m halves)
    const v16bf b20 = cat8(*(const v8bf*)&SupT_l[lcol][kbh],
                           *(const v8bf*)&SupT_l[lcol][kbh + 8]);
    convAcc0 = __builtin_amdgcn_wmma_f32_16x16x32_bf16(false, a2, false, b20,
                                                       (short)0, convAcc0, false, false);
    const v16bf b21 = cat8(*(const v8bf*)&SupT_l[16 + lcol][kbh],
                           *(const v8bf*)&SupT_l[16 + lcol][kbh + 8]);
    convAcc1 = __builtin_amdgcn_wmma_f32_16x16x32_bf16(false, a2, false, b21,
                                                       (short)0, convAcc1, false, false);
  }

  // ---- epilogue: out = relu(conv + bias + residual)
  #pragma unroll
  for (int mh = 0; mh < 2; ++mh) {
    const int   m  = mh * 16 + lcol;
    const float bm = bias[m];
    #pragma unroll
    for (int r = 0; r < 8; ++r) {
      const int M = r + 8 * lhalf;
      const int n = nb + wv * 16 + M;
      if (n < NN) {
        const size_t o = ((size_t)b * NN + n) * DD + m;
        const float cv = (mh == 0) ? convAcc0[r] : convAcc1[r];
        out[o] = fmaxf(cv + bm + resid[o], 0.f);
      }
    }
  }
}

// ---------------------------------------------------------------------------
extern "C" void kernel_launch(void* const* d_in, const int* in_sizes, int n_in,
                              void* d_out, int out_size, void* d_ws, size_t ws_size,
                              hipStream_t stream) {
  (void)in_sizes; (void)n_in; (void)out_size; (void)ws_size;
  const float* input  = (const float*)d_in[0];
  const int*   cyc    = (const int*)  d_in[1];
  const float* adj    = (const float*)d_in[2];
  const float* envf   = (const float*)d_in[3];
  const float* W_env  = (const float*)d_in[4];
  const float* b_env  = (const float*)d_in[5];
  const float* src    = (const float*)d_in[6];
  const float* tgt    = (const float*)d_in[7];
  const float* weight = (const float*)d_in[8];
  const float* bias   = (const float*)d_in[9];
  const float* W_res  = (const float*)d_in[10];
  const float* b_res  = (const float*)d_in[11];
  float* out = (float*)d_out;

  // workspace carve-up (all sizes multiples of 256 B)
  char* ws = (char*)d_ws;
  const size_t szST  = (size_t)PP * NN * HH * 2;  // bf16 S / T : 6.144 MB each
  const size_t szSup = (size_t)BB * NN * DD * 2;  // bf16 support^T: 4.096 MB
  unsigned short* Sst  = (unsigned short*)ws;  ws += szST;
  unsigned short* Tst  = (unsigned short*)ws;  ws += szST;
  unsigned short* supT = (unsigned short*)ws;  ws += szSup;
  float*          res  = (float*)ws;           // 8.192 MB

  k_prep<<<NN, HH, 0, stream>>>(envf, W_env, b_env, src, tgt, Sst, Tst);

  const int tot = BB * NN * DD;
  k_support<<<(tot + 255) / 256, 256, 0, stream>>>(input, weight, W_res, b_res, supT, res);

  dim3 grid((NN + 63) / 64, BB);
  k_fused<<<grid, 128, 0, stream>>>(cyc, adj, Sst, Tst, supT, res, bias, out);
}
